// SpikeDrivenTransformer_63565515980915
// MI455X (gfx1250) — compile-verified, matrix operand order checked by reading, not run on hardware
//
#include <hip/hip_runtime.h>

// ---- problem constants (from reference) ----
#define T_STEPS 4
#define BATCH   16
#define CHN     512
#define NCOLS   256                  // H*W
#define IMGS    (T_STEPS * BATCH)    // 64
#define HID     (4 * CHN)            // 2048
#define SPIKE_ELEMS ((size_t)IMGS * NCOLS * CHN)   // 8,388,608
#define HID_ELEMS   ((size_t)IMGS * NCOLS * HID)   // 33,554,432

#define KCHUNK   128                 // K depth staged per LDS round
#define BCOLS    32                  // N columns staged per workgroup
#define LDS_STRIDE (KCHUNK + 8)      // +8 bf16 pad -> 272B col stride (4-bank step)

typedef __attribute__((ext_vector_type(16))) __bf16       v16bf;
typedef __attribute__((ext_vector_type(8)))  float        v8f;
typedef __attribute__((ext_vector_type(4)))  unsigned int u32x4;
typedef __attribute__((ext_vector_type(4)))  int          i32x4;

union FragBF { v16bf v; u32x4 q[2]; };

#if defined(__gfx1250__) && \
    __has_builtin(__builtin_amdgcn_global_load_async_to_lds_b128) && \
    __has_builtin(__builtin_amdgcn_s_wait_asynccnt)
#define USE_ASYNC_LDS 1
typedef __attribute__((address_space(1))) i32x4 as1_i32x4;
typedef __attribute__((address_space(3))) i32x4 as3_i32x4;
#endif

__device__ __forceinline__ __bf16 f2bf(float f) {
  union { float f; unsigned u; } c; c.f = f;
  unsigned r = c.u + 0x7FFFu + ((c.u >> 16) & 1u);   // round-to-nearest-even
  unsigned short h = (unsigned short)(r >> 16);
  return __builtin_bit_cast(__bf16, h);
}
__device__ __forceinline__ float bf2f(__bf16 b) {
  unsigned short h = __builtin_bit_cast(unsigned short, b);
  union { unsigned u; float f; } c; c.u = ((unsigned)h) << 16;
  return c.f;
}

// ---------- small prep kernels ----------
__global__ void cvt_bf16_kernel(const float* __restrict__ in, __bf16* __restrict__ out, int n) {
  int e = blockIdx.x * blockDim.x + threadIdx.x;
  if (e < n) out[e] = f2bf(in[e]);
}

// bnp: [4][C] = gamma, beta, mean, var.  scale = g/sqrt(v+eps),
// shift = (bias - mean)*scale + beta   (bias may be null)
__global__ void bn_prep_kernel(const float* __restrict__ bnp, const float* __restrict__ bias,
                               float* __restrict__ scale, float* __restrict__ shift, int C) {
  int c = blockIdx.x * blockDim.x + threadIdx.x;
  if (c >= C) return;
  float g  = bnp[c];
  float be = bnp[C + c];
  float m  = bnp[2 * C + c];
  float v  = bnp[3 * C + c];
  float sc = g * rsqrtf(v + 1e-5f);
  float b  = bias ? bias[c] : 0.0f;
  scale[c] = sc;
  shift[c] = (b - m) * sc + be;
}

// ---------- LIF kernels (TAU=2, hard reset to 0) ----------
// input x: f32 [T][B][C][N]  ->  spikes bf16 [T*B][N][C]  (layout transpose for GEMM)
__global__ void lif_input_kernel(const float* __restrict__ x, __bf16* __restrict__ xs) {
  const int E = BATCH * CHN * NCOLS;
  int e = blockIdx.x * blockDim.x + threadIdx.x;
  if (e >= E) return;
  int n = e % NCOLS;
  int c = (e / NCOLS) % CHN;
  int b = e / (NCOLS * CHN);
  float v = 0.0f;
  for (int t = 0; t < T_STEPS; ++t) {
    float xt = x[(size_t)t * E + e];
    v += (xt - v) * 0.5f;
    float s = (v >= 1.0f) ? 1.0f : 0.0f;
    v *= (1.0f - s);
    xs[(((size_t)(t * BATCH + b)) * NCOLS + n) * CHN + c] = f2bf(s);
  }
}

// in-place temporal LIF on bf16 buffer laid out [T][E]
__global__ void lif_bf16_inplace_kernel(__bf16* __restrict__ buf, int E, float vth) {
  int e = blockIdx.x * blockDim.x + threadIdx.x;
  if (e >= E) return;
  float v = 0.0f;
  for (int t = 0; t < T_STEPS; ++t) {
    float xt = bf2f(buf[(size_t)t * E + e]);
    v += (xt - v) * 0.5f;
    float s = (v >= vth) ? 1.0f : 0.0f;
    v *= (1.0f - s);
    buf[(size_t)t * E + e] = f2bf(s);
  }
}

// f32 [T][E] -> bf16 spikes [T][E]
__global__ void lif_f32_bf16_kernel(const float* __restrict__ in, __bf16* __restrict__ out,
                                    int E, float vth) {
  int e = blockIdx.x * blockDim.x + threadIdx.x;
  if (e >= E) return;
  float v = 0.0f;
  for (int t = 0; t < T_STEPS; ++t) {
    float xt = in[(size_t)t * E + e];
    v += (xt - v) * 0.5f;
    float s = (v >= vth) ? 1.0f : 0.0f;
    v *= (1.0f - s);
    out[(size_t)t * E + e] = f2bf(s);
  }
}

// ---------- linear-attention helpers ----------
// kv[img][c] = sum_n k[img][n][c] * v[img][n][c]
__global__ void kv_reduce_kernel(const __bf16* __restrict__ ks, const __bf16* __restrict__ vs,
                                 float* __restrict__ kvpre) {
  int e = blockIdx.x * blockDim.x + threadIdx.x;   // IMGS*CHN = 32768
  if (e >= IMGS * CHN) return;
  int c   = e % CHN;
  int img = e / CHN;
  const __bf16* kp = ks + (size_t)img * NCOLS * CHN + c;
  const __bf16* vp = vs + (size_t)img * NCOLS * CHN + c;
  float s = 0.0f;
  for (int n = 0; n < NCOLS; ++n)
    s += bf2f(kp[(size_t)n * CHN]) * bf2f(vp[(size_t)n * CHN]);
  kvpre[e] = s;
}

// att[img][n][c] = q_spike * kv_spike (broadcast over n)
__global__ void att_mul_kernel(const __bf16* __restrict__ qs, const __bf16* __restrict__ kvs,
                               __bf16* __restrict__ att) {
  size_t e = (size_t)blockIdx.x * blockDim.x + threadIdx.x;
  if (e >= SPIKE_ELEMS) return;
  int c    = (int)(e % CHN);
  int img  = (int)(e / ((size_t)CHN * NCOLS));
  att[e] = f2bf(bf2f(qs[e]) * bf2f(kvs[(size_t)img * CHN + c]));
}

// vh output: vs [img][n][c] bf16 -> f32 [img][h][n][ch], c = h*64+ch
__global__ void vh_copy_kernel(const __bf16* __restrict__ vs, float* __restrict__ outVh) {
  size_t e = (size_t)blockIdx.x * blockDim.x + threadIdx.x;
  if (e >= SPIKE_ELEMS) return;
  int c   = (int)(e % CHN);
  int n   = (int)((e / CHN) % NCOLS);
  int img = (int)(e / ((size_t)CHN * NCOLS));
  int h = c >> 6, ch = c & 63;
  outVh[((((size_t)img * 8 + h) * NCOLS) + n) * 64 + ch] = bf2f(vs[e]);
}

// ---------- WMMA GEMM:  Y[img] = epilogue(W[M,K] x X[img][K,Ncols]) ----------
// X stored K-contiguous per column: X[img][n][K].
// Workgroup (8 waves) computes a 128(M) x 32(N) macro-tile per image:
//   - B panel (32 cols x 128 K) staged in LDS per K-chunk, shared by all waves
//     (async global->LDS when the builtin exists; global+ds_store fallback)
//   - each wave: one M-tile x two N-tiles, A fragment reused across both
// mode 0: out bf16 [img][n][M]   = acc*scale + shift                        (pre-act)
// mode 1: out f32  [img][n][M]   = acc*scale + shift + ident[img][m][n]     (proj+identity)
// mode 2: out f32  [img][m][n]   = acc*scale + shift + ident[img][n][m]     (fc2+residual)
__global__ void __launch_bounds__(256)
wmma_gemm_kernel(const __bf16* __restrict__ Wm, const __bf16* __restrict__ X,
                 const float* __restrict__ scale, const float* __restrict__ shift,
                 const float* __restrict__ ident, void* __restrict__ out,
                 int M, int K, int mode) {
  __shared__ __align__(16) __bf16 sB[BCOLS * LDS_STRIDE];   // 32 * 272B = 8.5KB

  const int tid  = threadIdx.x;
  const int lane = tid & 31;
  const int wave = tid >> 5;

  const int nGroups = NCOLS / BCOLS;            // 8
  const int mGroups = M >> 7;                   // M/128
  int bid = blockIdx.x;
  const int img = bid / (nGroups * mGroups);
  int rem = bid - img * (nGroups * mGroups);
  const int mG = rem / nGroups;
  const int nG = rem - mG * nGroups;
  const int m0 = (mG * 8 + wave) << 4;
  const int n0 = nG * BCOLS;

  // bf16 16x32 A fragment: lane<16 -> row=lane,    K {0..7, 16..23}
  //                        lane>=16 -> row=lane-16, K {8..15, 24..31}
  const int kSel = (lane & 16) ? 8 : 0;
  const __bf16* Ap = Wm + (size_t)(m0 + (lane & 15)) * K + kSel;
  const __bf16* Xc = X + ((size_t)img * NCOLS + n0) * K;   // base of 32-col panel

  v8f accs[2] = {};
  for (int kc = 0; kc < K; kc += KCHUNK) {
    // ---- stage B panel: 32 cols x 128 K = 8KB = 512 x 16B, 2 per thread ----
#pragma unroll
    for (int s = 0; s < 2; ++s) {
      int idx = tid + s * 256;                  // 0..511
      int col = idx >> 4;                       // 0..31
      int ck  = (idx & 15) << 3;                // k offset in elems (16B chunks)
      const __bf16* g = Xc + (size_t)col * K + kc + ck;
      __bf16* l = sB + col * LDS_STRIDE + ck;
#if defined(USE_ASYNC_LDS)
      __builtin_amdgcn_global_load_async_to_lds_b128(
          (as1_i32x4*)(__bf16*)g, (as3_i32x4*)l, 0, 0);
#else
      *(u32x4*)l = *(const u32x4*)g;
#endif
    }
#if defined(USE_ASYNC_LDS)
    __builtin_amdgcn_s_wait_asynccnt(0);
#endif
    __syncthreads();

    __builtin_prefetch(Ap + kc + KCHUNK, 0, 1);          // next A chunk hint
#pragma unroll
    for (int j = 0; j < KCHUNK; j += 32) {
      FragBF a, b0, b1;
      a.q[0] = *(const u32x4*)(Ap + kc + j);
      a.q[1] = *(const u32x4*)(Ap + kc + j + 16);
      const __bf16* lb0 = sB + (lane & 15) * LDS_STRIDE + j + kSel;
      const __bf16* lb1 = lb0 + 16 * LDS_STRIDE;
      b0.q[0] = *(const u32x4*)lb0;
      b0.q[1] = *(const u32x4*)(lb0 + 16);
      b1.q[0] = *(const u32x4*)lb1;
      b1.q[1] = *(const u32x4*)(lb1 + 16);
      accs[0] = __builtin_amdgcn_wmma_f32_16x16x32_bf16(false, a.v, false, b0.v,
                                                        (short)0, accs[0], false, false);
      accs[1] = __builtin_amdgcn_wmma_f32_16x16x32_bf16(false, a.v, false, b1.v,
                                                        (short)0, accs[1], false, false);
    }
    __syncthreads();
  }

  // D layout: lane<16 -> N=lane, M=m0+r ; lane>=16 -> N=lane-16, M=m0+8+r
  const int mBase = m0 + ((lane & 16) ? 8 : 0);
#pragma unroll
  for (int half = 0; half < 2; ++half) {
    const int nOut = n0 + half * 16 + (lane & 15);
#pragma unroll
    for (int r = 0; r < 8; ++r) {
      int m = mBase + r;
      float z = accs[half][r] * scale[m] + shift[m];
      if (mode == 0) {
        ((__bf16*)out)[((size_t)img * NCOLS + nOut) * M + m] = f2bf(z);
      } else if (mode == 1) {
        z += ident[((size_t)img * M + m) * NCOLS + nOut];
        ((float*)out)[((size_t)img * NCOLS + nOut) * M + m] = z;
      } else {
        z += ident[((size_t)img * NCOLS + nOut) * M + m];
        ((float*)out)[((size_t)img * M + m) * NCOLS + nOut] = z;
      }
    }
  }
}

// ---------- host launcher ----------
extern "C" void kernel_launch(void* const* d_in, const int* in_sizes, int n_in,
                              void* d_out, int out_size, void* d_ws, size_t ws_size,
                              hipStream_t stream) {
  (void)in_sizes; (void)n_in; (void)out_size; (void)ws_size;

  const float* x       = (const float*)d_in[0];
  const float* q_w     = (const float*)d_in[1];
  const float* k_w     = (const float*)d_in[2];
  const float* v_w     = (const float*)d_in[3];
  const float* proj_w  = (const float*)d_in[4];
  const float* proj_b  = (const float*)d_in[5];
  const float* fc1_w   = (const float*)d_in[6];
  const float* fc1_b   = (const float*)d_in[7];
  const float* fc2_w   = (const float*)d_in[8];
  const float* fc2_b   = (const float*)d_in[9];
  const float* q_bn    = (const float*)d_in[10];
  const float* k_bn    = (const float*)d_in[11];
  const float* v_bn    = (const float*)d_in[12];
  const float* proj_bn = (const float*)d_in[13];
  const float* fc1_bn  = (const float*)d_in[14];
  const float* fc2_bn  = (const float*)d_in[15];

  float* outMain = (float*)d_out;                  // [T,B,C,H,W]
  float* outVh   = outMain + SPIKE_ELEMS;          // [T,B,h,N,Ch]

  // workspace carve-up
  char* base = (char*)d_ws;
  size_t off = 0;
  auto take = [&](size_t bytes) -> char* {
    char* p = base + off;
    off += (bytes + 255) & ~(size_t)255;
    return p;
  };
  __bf16* xs    = (__bf16*)take(SPIKE_ELEMS * 2);  // shortcut spikes; reused as att
  __bf16* qs    = (__bf16*)take(SPIKE_ELEMS * 2);  // q pre -> q spikes; reused as h1
  __bf16* ks    = (__bf16*)take(SPIKE_ELEMS * 2);
  __bf16* vs    = (__bf16*)take(SPIKE_ELEMS * 2);
  float*  xattn = (float*) take(SPIKE_ELEMS * 4);
  __bf16* hbuf  = (__bf16*)take(HID_ELEMS * 2);    // fc1 pre -> h2 spikes (in place)
  __bf16* qwb   = (__bf16*)take((size_t)CHN * CHN * 2);
  __bf16* kwb   = (__bf16*)take((size_t)CHN * CHN * 2);
  __bf16* vwb   = (__bf16*)take((size_t)CHN * CHN * 2);
  __bf16* pwb   = (__bf16*)take((size_t)CHN * CHN * 2);
  __bf16* f1wb  = (__bf16*)take((size_t)HID * CHN * 2);
  __bf16* f2wb  = (__bf16*)take((size_t)CHN * HID * 2);
  float*  qsc = (float*)take(CHN * 4);  float* qsh = (float*)take(CHN * 4);
  float*  ksc = (float*)take(CHN * 4);  float* ksh = (float*)take(CHN * 4);
  float*  vsc = (float*)take(CHN * 4);  float* vsh = (float*)take(CHN * 4);
  float*  psc = (float*)take(CHN * 4);  float* psh = (float*)take(CHN * 4);
  float*  f1sc = (float*)take(HID * 4); float* f1sh = (float*)take(HID * 4);
  float*  f2sc = (float*)take(CHN * 4); float* f2sh = (float*)take(CHN * 4);
  float*  kvpre = (float*) take((size_t)IMGS * CHN * 4);
  __bf16* kvs   = (__bf16*)take((size_t)IMGS * CHN * 2);

  auto cvt = [&](const float* in, __bf16* o, int n) {
    cvt_bf16_kernel<<<(n + 255) / 256, 256, 0, stream>>>(in, o, n);
  };
  auto bnp = [&](const float* p, const float* bias, float* sc, float* sh, int C) {
    bn_prep_kernel<<<(C + 255) / 256, 256, 0, stream>>>(p, bias, sc, sh, C);
  };
  auto gemm = [&](const __bf16* Wm, const __bf16* X, const float* sc, const float* sh,
                  const float* ident, void* out, int M, int K, int mode) {
    int blocks = IMGS * (NCOLS / BCOLS) * (M / 128);
    wmma_gemm_kernel<<<blocks, 256, 0, stream>>>(Wm, X, sc, sh, ident, out, M, K, mode);
  };

  // 0) weight + BN prep
  cvt(q_w, qwb, CHN * CHN);  cvt(k_w, kwb, CHN * CHN);  cvt(v_w, vwb, CHN * CHN);
  cvt(proj_w, pwb, CHN * CHN);
  cvt(fc1_w, f1wb, HID * CHN);  cvt(fc2_w, f2wb, CHN * HID);
  bnp(q_bn, nullptr, qsc, qsh, CHN);
  bnp(k_bn, nullptr, ksc, ksh, CHN);
  bnp(v_bn, nullptr, vsc, vsh, CHN);
  bnp(proj_bn, proj_b, psc, psh, CHN);
  bnp(fc1_bn, fc1_b, f1sc, f1sh, HID);
  bnp(fc2_bn, fc2_b, f2sc, f2sh, CHN);

  const int E_bcn = BATCH * CHN * NCOLS;       // 2,097,152 (per-timestep elems, C=512)
  const int E_hid = BATCH * HID * NCOLS;       // 8,388,608 (per-timestep elems, 4C)

  // 1) shortcut LIF (sigmoid surrogate, vth=1): x -> xs spikes [img][n][C]
  lif_input_kernel<<<E_bcn / 256, 256, 0, stream>>>(x, xs);

  // 2) q/k/v = LIF(BN(conv1x1(xs)))
  gemm(qwb, xs, qsc, qsh, nullptr, qs, CHN, CHN, 0);
  gemm(kwb, xs, ksc, ksh, nullptr, ks, CHN, CHN, 0);
  gemm(vwb, xs, vsc, vsh, nullptr, vs, CHN, CHN, 0);
  lif_bf16_inplace_kernel<<<E_bcn / 256, 256, 0, stream>>>(qs, E_bcn, 1.0f);
  lif_bf16_inplace_kernel<<<E_bcn / 256, 256, 0, stream>>>(ks, E_bcn, 1.0f);
  lif_bf16_inplace_kernel<<<E_bcn / 256, 256, 0, stream>>>(vs, E_bcn, 1.0f);

  // 3) vh output (second tuple element)
  vh_copy_kernel<<<(unsigned)(SPIKE_ELEMS / 256), 256, 0, stream>>>(vs, outVh);

  // 4) linear attention: kv = LIF_{0.5}(sum_n k*v); att = q * kv  (att reuses xs)
  kv_reduce_kernel<<<(IMGS * CHN) / 256, 256, 0, stream>>>(ks, vs, kvpre);
  lif_f32_bf16_kernel<<<(BATCH * CHN) / 256, 256, 0, stream>>>(kvpre, kvs, BATCH * CHN, 0.5f);
  att_mul_kernel<<<(unsigned)(SPIKE_ELEMS / 256), 256, 0, stream>>>(qs, kvs, xs);

  // 5) x_attn = BN(proj(att)) + identity(x)
  gemm(pwb, xs, psc, psh, x, xattn, CHN, CHN, 1);

  // 6) MLP: h1 = LIF(x_attn); fc1+BN; LIF; fc2+BN + x_attn -> d_out
  lif_f32_bf16_kernel<<<E_bcn / 256, 256, 0, stream>>>(xattn, qs, E_bcn, 1.0f);
  gemm(f1wb, qs, f1sc, f1sh, nullptr, hbuf, HID, CHN, 0);
  lif_bf16_inplace_kernel<<<E_hid / 256, 256, 0, stream>>>(hbuf, E_hid, 1.0f);
  gemm(f2wb, hbuf, f2sc, f2sh, xattn, outMain, CHN, HID, 2);
}
